// YOLOv11Loss_48352741818725
// MI455X (gfx1250) — compile-verified, matrix operand order checked by reading, not run on hardware
//
#include <hip/hip_runtime.h>
#include <hip/hip_bf16.h>
#include <math.h>

typedef __attribute__((ext_vector_type(16))) _Float16 v16h_t;
typedef __attribute__((ext_vector_type(8)))  float    v8f_t;

#define BATCH    8
#define NANCH    8400
#define MAXGT    8
#define TOPK     10
#define KSEL     80
#define REGMAX   16
#define NPROTO   32
#define HP       160
#define HWPROTO  (HP*HP)      // 25600
#define MT       5            // 80/16 row tiles
#define NT       1600         // 25600/16 col tiles
#define TPW      4            // column tiles per wave in mask kernel
#define EPSF     1e-9f

__device__ __forceinline__ void decode_anchor(int n, int& lvl, int& w, int& i,
                                              float& stride, float& ax, float& ay) {
    if (n < 6400)      { lvl = 0; w = 80; stride = 8.f;  i = n; }
    else if (n < 8000) { lvl = 1; w = 40; stride = 16.f; i = n - 6400; }
    else               { lvl = 2; w = 20; stride = 32.f; i = n - 8000; }
    int y = i / w, x = i - y * w;
    ax = (x + 0.5f) * stride;
    ay = (y + 0.5f) * stride;
}

__device__ __forceinline__ float bce_logits(float x, float t) {
    return fmaxf(x, 0.f) - x * t + log1pf(expf(-fabsf(x)));
}

// fast version for the 16.4M-element mask BCE (hardware v_exp_f32/v_log_f32)
__device__ __forceinline__ float bce_logits_fast(float x, float t) {
    return fmaxf(x, 0.f) - x * t + __logf(1.f + __expf(-fabsf(x)));
}

// ---------------- Stage A: decode boxes + scores ----------------
__global__ void stage_decode(const float* __restrict__ box3, const float* __restrict__ box4,
                             const float* __restrict__ box5, const float* __restrict__ cls3,
                             const float* __restrict__ cls4, const float* __restrict__ cls5,
                             float* __restrict__ boxes, float* __restrict__ scores) {
    int t = blockIdx.x * blockDim.x + threadIdx.x;
    if (t >= BATCH * NANCH) return;
    int b = t / NANCH, n = t - b * NANCH;
    int lvl, w, i; float s, ax, ay;
    decode_anchor(n, lvl, w, i, s, ax, ay);
    const float* bp = (lvl == 0) ? box3 : (lvl == 1) ? box4 : box5;
    const float* cp = (lvl == 0) ? cls3 : (lvl == 1) ? cls4 : cls5;
    int hw = w * w;
    float d[4];
    for (int side = 0; side < 4; ++side) {
        float lg[REGMAX], m = -1e30f;
        for (int j = 0; j < REGMAX; ++j) {
            lg[j] = bp[((size_t)b * 64 + side * REGMAX + j) * hw + i];
            m = fmaxf(m, lg[j]);
        }
        float se = 0.f, dot = 0.f;
        for (int j = 0; j < REGMAX; ++j) {
            float e = expf(lg[j] - m);
            se += e; dot += e * (float)j;
        }
        d[side] = (dot / se) * s;
    }
    float* ob = boxes + (size_t)t * 4;
    ob[0] = ax - d[0]; ob[1] = ay - d[1]; ob[2] = ax + d[2]; ob[3] = ay + d[3];
    float cl = cp[(size_t)b * hw + i];
    scores[t] = 1.f / (1.f + expf(-cl));
}

// ---------------- Stage B: assignment (one block per batch) ----------------
__global__ void __launch_bounds__(256)
stage_assign(const float* __restrict__ boxes, const float* __restrict__ scores,
             const float* __restrict__ gtb, float* __restrict__ metric,
             int* __restrict__ selIdx, int* __restrict__ selGi,
             float* __restrict__ selVf, float* __restrict__ accum) {
    int b = blockIdx.x, tid = threadIdx.x;
    __shared__ float sgt[MAXGT * 4];
    __shared__ unsigned char mpos[NANCH];
    __shared__ unsigned char bestG[NANCH];
    __shared__ float redV[256];
    __shared__ int   redI[256];
    if (tid < MAXGT * 4) sgt[tid] = gtb[b * MAXGT * 4 + tid];
    __syncthreads();

    for (int n = tid; n < NANCH; n += 256) {
        int lvl, w, i; float s, ax, ay;
        decode_anchor(n, lvl, w, i, s, ax, ay);
        const float* pb = boxes + ((size_t)b * NANCH + n) * 4;
        float px0 = pb[0], py0 = pb[1], px1 = pb[2], py1 = pb[3];
        float ap = fmaxf(px1 - px0, 0.f) * fmaxf(py1 - py0, 0.f);
        float sc = scores[b * NANCH + n];
        float bv = -1e30f; int bg = 0;
        for (int g = 0; g < MAXGT; ++g) {
            float gx0 = sgt[g*4], gy0 = sgt[g*4+1], gx1 = sgt[g*4+2], gy1 = sgt[g*4+3];
            float ix = fmaxf(fminf(gx1, px1) - fmaxf(gx0, px0), 0.f);
            float iy = fmaxf(fminf(gy1, py1) - fmaxf(gy0, py0), 0.f);
            float inter = ix * iy;
            float ag = fmaxf(gx1 - gx0, 0.f) * fmaxf(gy1 - gy0, 0.f);
            float iou = inter / (ag + ap - inter + EPSF);
            if (iou > bv) { bv = iou; bg = g; }
            float dmin = fminf(fminf(ax - gx0, ay - gy0), fminf(gx1 - ax, gy1 - ay));
            float i2 = iou * iou; float i6 = i2 * i2 * i2;
            metric[((size_t)(b * MAXGT + g)) * NANCH + n] = (dmin > EPSF) ? sc * i6 : 0.f;
        }
        bestG[n] = (unsigned char)bg;
        mpos[n] = 0;
    }
    __syncthreads();

    for (int g = 0; g < MAXGT; ++g) {
        float* mb = metric + ((size_t)(b * MAXGT + g)) * NANCH;
        for (int it = 0; it < TOPK; ++it) {
            float bv = -1e30f; int bi = NANCH;
            for (int n = tid; n < NANCH; n += 256) {
                float v = mb[n];
                if (v > bv || (v == bv && n < bi)) { bv = v; bi = n; }
            }
            redV[tid] = bv; redI[tid] = bi;
            __syncthreads();
            for (int s = 128; s > 0; s >>= 1) {
                if (tid < s) {
                    float ov = redV[tid + s]; int oi = redI[tid + s];
                    if (ov > redV[tid] || (ov == redV[tid] && oi < redI[tid])) {
                        redV[tid] = ov; redI[tid] = oi;
                    }
                }
                __syncthreads();
            }
            if (tid == 0) {
                int idx = redI[0];
                if (idx < NANCH) {
                    if (redV[0] > EPSF) mpos[idx] |= (unsigned char)(1u << g);
                    mb[idx] = -1.f;
                }
            }
            __threadfence_block();
            __syncthreads();
        }
    }

    for (int n = tid; n < NANCH; n += 256) {
        unsigned int m = mpos[n];
        if (__popc(m) > 1) m = 1u << bestG[n];
        mpos[n] = (unsigned char)m;
    }
    __syncthreads();

    if (tid == 0) {
        int cnt = 0;
        for (int n = 0; n < NANCH; ++n) {
            unsigned int m = mpos[n];
            if (m) {
                if (cnt < KSEL) {
                    selIdx[b * KSEL + cnt] = n;
                    selGi[b * KSEL + cnt]  = __ffs(m) - 1;
                    selVf[b * KSEL + cnt]  = 1.f;
                }
                ++cnt;
            }
        }
        int npos = cnt < KSEL ? cnt : KSEL;
        for (int k = npos; k < KSEL; ++k) {
            selIdx[b * KSEL + k] = 0; selGi[b * KSEL + k] = 0; selVf[b * KSEL + k] = 0.f;
        }
        accum[b * 8 + 0] = 0.f; accum[b * 8 + 1] = 0.f;
        accum[b * 8 + 2] = 0.f; accum[b * 8 + 3] = 0.f;
        accum[b * 8 + 4] = (float)npos;
    }
}

// ---------------- Stage C: box / cls / dfl losses ----------------
__global__ void stage_posloss(const float* __restrict__ boxes, const float* __restrict__ gtb,
                              const float* __restrict__ box3, const float* __restrict__ box4,
                              const float* __restrict__ box5, const float* __restrict__ cls3,
                              const float* __restrict__ cls4, const float* __restrict__ cls5,
                              const int* __restrict__ selIdx, const int* __restrict__ selGi,
                              const float* __restrict__ selVf, float* __restrict__ accum) {
    int t = blockIdx.x * blockDim.x + threadIdx.x;
    if (t >= BATCH * KSEL) return;
    int b = t / KSEL;
    float vf = selVf[t];
    int n = selIdx[t], gi = selGi[t];
    int lvl, w, i; float s, ax, ay;
    decode_anchor(n, lvl, w, i, s, ax, ay);
    const float* pb = boxes + ((size_t)b * NANCH + n) * 4;
    const float* gb = gtb + (b * MAXGT + gi) * 4;
    float ix = fmaxf(fminf(gb[2], pb[2]) - fmaxf(gb[0], pb[0]), 0.f);
    float iy = fmaxf(fminf(gb[3], pb[3]) - fmaxf(gb[1], pb[1]), 0.f);
    float inter = ix * iy;
    float apr = (pb[2] - pb[0]) * (pb[3] - pb[1]);
    float agt = (gb[2] - gb[0]) * (gb[3] - gb[1]);
    float iou = inter / (apr + agt - inter + EPSF);
    float lb = (1.f - iou) * vf;

    const float* cp = (lvl == 0) ? cls3 : (lvl == 1) ? cls4 : cls5;
    int hw = w * w;
    float cl = cp[(size_t)b * hw + i];
    float tgt = fminf(fmaxf(iou, 0.f), 1.f);
    float lc = bce_logits(cl, tgt) * vf;

    const float* bp = (lvl == 0) ? box3 : (lvl == 1) ? box4 : box5;
    float ltrb[4] = { (ax - gb[0]) / s, (ay - gb[1]) / s, (gb[2] - ax) / s, (gb[3] - ay) / s };
    float ld = 0.f;
    for (int side = 0; side < 4; ++side) {
        float tt = fminf(fmaxf(ltrb[side], 0.f), (float)REGMAX - 1e-6f);
        int tl = (int)tt;
        bool atb = tl >= REGMAX - 1;
        int tr = atb ? tl : min(tl + 1, REGMAX - 1);
        float wr = atb ? 0.f : tt - (float)tl;
        float wl = 1.f - wr;
        float lg[REGMAX], m = -1e30f;
        for (int j = 0; j < REGMAX; ++j) {
            lg[j] = bp[((size_t)b * 64 + side * REGMAX + j) * hw + i];
            m = fmaxf(m, lg[j]);
        }
        float se = 0.f;
        for (int j = 0; j < REGMAX; ++j) se += expf(lg[j] - m);
        float logZ = m + logf(se);
        ld += (logZ - lg[tl]) * wl + (logZ - lg[tr]) * wr;
    }
    ld *= vf;
    atomicAdd(&accum[b * 8 + 0], lc);
    atomicAdd(&accum[b * 8 + 1], lb);
    atomicAdd(&accum[b * 8 + 2], ld);
}

// ---------------- Stage D1: bilinear downsample gt_masks 640 -> 160 ----------------
__global__ void stage_downsample(const float* __restrict__ gtm, float* __restrict__ dm) {
    int t = blockIdx.x * blockDim.x + threadIdx.x;
    if (t >= BATCH * MAXGT * HWPROTO) return;
    int p = t % HWPROTO;
    int bg = t / HWPROTO;
    int yo = p / HP, xo = p - yo * HP;
    float sy = (yo + 0.5f) * 4.f - 0.5f; sy = fminf(fmaxf(sy, 0.f), 639.f);
    float sx = (xo + 0.5f) * 4.f - 0.5f; sx = fminf(fmaxf(sx, 0.f), 639.f);
    int y0 = (int)sy, x0 = (int)sx;
    int y1 = min(y0 + 1, 639), x1 = min(x0 + 1, 639);
    float wy = sy - y0, wx = sx - x0;
    const float* base = gtm + (size_t)bg * 640 * 640;
    float a = base[y0 * 640 + x0], bb = base[y0 * 640 + x1];
    float c = base[y1 * 640 + x0], d = base[y1 * 640 + x1];
    float top = a * (1.f - wx) + bb * wx;
    float bot = c * (1.f - wx) + d * wx;
    dm[t] = top * (1.f - wy) + bot * wy;
}

// ---------------- Stage D2: swizzle coef[sel] into WMMA 16x32 f16 A layout ----------------
__global__ void stage_swizzleA(const float* __restrict__ coef3, const float* __restrict__ coef4,
                               const float* __restrict__ coef5, const int* __restrict__ selIdx,
                               _Float16* __restrict__ Asw) {
    int t = blockIdx.x * blockDim.x + threadIdx.x;   // B*MT*32 = 1280 threads
    if (t >= BATCH * MT * 32) return;
    int lane = t % 32;
    int bm = t / 32;
    int b = bm / MT, mt = bm - b * MT;
    int row = mt * 16 + (lane & 15);
    int n = selIdx[b * KSEL + row];
    int lvl, w, i; float s, ax, ay;
    decode_anchor(n, lvl, w, i, s, ax, ay);
    const float* cp = (lvl == 0) ? coef3 : (lvl == 1) ? coef4 : coef5;
    int hw = w * w;
    _Float16* out = Asw + (size_t)t * 16;
#pragma unroll
    for (int e = 0; e < 16; ++e) {
        int kk = e + ((e < 8) ? 0 : 8) + ((lane < 16) ? 0 : 8);   // ISA 16-bit A 16x32 layout
        out[e] = (_Float16)cp[((size_t)b * NPROTO + kk) * hw + i];
    }
}

// ---------------- Stage D3: convert + swizzle proto into WMMA 32x16 f16 B layout ----------
__global__ void stage_swizzleB(const float* __restrict__ proto, _Float16* __restrict__ Bsw) {
    int t = blockIdx.x * blockDim.x + threadIdx.x;   // B*NT*32 = 409600 threads
    if (t >= BATCH * NT * 32) return;
    int lane = t & 31;
    int bn = t >> 5;
    int b = bn / NT, nt = bn - b * NT;
    int p = nt * 16 + (lane & 15);
    const float* src = proto + (size_t)b * NPROTO * HWPROTO + p;
    _Float16* out = Bsw + (size_t)t * 16;
#pragma unroll
    for (int e = 0; e < 16; ++e) {
        int kk = e + ((e < 8) ? 0 : 8) + ((lane < 16) ? 0 : 8);   // mirrored A layout (B cols)
        out[e] = (_Float16)src[(size_t)kk * HWPROTO];
    }
}

// ---------------- Stage E: WMMA coef@proto tiles + fused mask BCE ----------------
__global__ void __launch_bounds__(256)
stage_maskloss(const _Float16* __restrict__ Bsw, const _Float16* __restrict__ Asw,
               const float* __restrict__ dm, const int* __restrict__ selGi,
               const float* __restrict__ selVf, float* __restrict__ accum) {
    int lane = threadIdx.x & 31;
    int wv = blockIdx.x * (blockDim.x >> 5) + (threadIdx.x >> 5);  // 16000 waves
    const int tilesPerB = MT * (NT / TPW);                          // 2000
    int b = wv / tilesPerB;
    int rem = wv - b * tilesPerB;
    int mt = rem / (NT / TPW);
    int ntBase = (rem - mt * (NT / TPW)) * TPW;

    // A tile: one contiguous 32B v16h load per lane
    v16h_t a = *(const v16h_t*)(Asw + ((size_t)((b * MT + mt) * 32 + lane)) * 16);

    // per-row metadata (loaded once, reused for all TPW column tiles)
    float vf[8]; int gi[8];
#pragma unroll
    for (int r = 0; r < 8; ++r) {
        int kr = mt * 16 + r + ((lane >= 16) ? 8 : 0);
        vf[r] = selVf[b * KSEL + kr];
        gi[r] = selGi[b * KSEL + kr];
    }

    float acc = 0.f;
#pragma unroll
    for (int u = 0; u < TPW; ++u) {
        int nt = ntBase + u;
        const _Float16* bp = Bsw + ((size_t)((b * NT + nt) * 32 + lane)) * 16;
        __builtin_prefetch(bp + 32 * 16, 0, 1);      // gfx1250 global_prefetch_b8
        v16h_t bm = *(const v16h_t*)bp;
        v8f_t c = {};
        c = __builtin_amdgcn_wmma_f32_16x16x32_f16(false, a, false, bm,
                                                   (short)0, c, false, false);
        int p = nt * 16 + (lane & 15);
#pragma unroll
        for (int r = 0; r < 8; ++r) {
            float tm = dm[((size_t)(b * MAXGT + gi[r])) * HWPROTO + p];
            acc += bce_logits_fast(c[r], tm) * vf[r];
        }
    }
    acc *= 1.f / (float)HWPROTO;
    for (int off = 16; off > 0; off >>= 1) acc += __shfl_xor(acc, off, 32);
    if (lane == 0) atomicAdd(&accum[b * 8 + 3], acc);
}

// ---------------- Stage F: final weighted reduction ----------------
__global__ void stage_final(const float* __restrict__ accum, float* __restrict__ out) {
    if (threadIdx.x == 0 && blockIdx.x == 0) {
        float lc = 0, lb = 0, ld = 0, lm = 0, np = 0;
        for (int b = 0; b < BATCH; ++b) {
            lc += accum[b * 8 + 0]; lb += accum[b * 8 + 1];
            ld += accum[b * 8 + 2]; lm += accum[b * 8 + 3];
            np += accum[b * 8 + 4];
        }
        np = fmaxf(np, 1.f);
        float loss_cls = lc / np;
        float loss_box = lb / np;
        float loss_dfl = ld / (np * 4.f);
        float loss_mask = lm / np;
        out[0] = 7.5f * loss_box + 0.5f * loss_cls + 2.5f * loss_mask + 1.5f * loss_dfl;
        out[1] = loss_cls; out[2] = loss_box; out[3] = loss_dfl; out[4] = loss_mask;
    }
}

extern "C" void kernel_launch(void* const* d_in, const int* in_sizes, int n_in,
                              void* d_out, int out_size, void* d_ws, size_t ws_size,
                              hipStream_t stream) {
    const float* box3  = (const float*)d_in[0];
    const float* cls3  = (const float*)d_in[1];
    const float* coef3 = (const float*)d_in[2];
    const float* box4  = (const float*)d_in[3];
    const float* cls4  = (const float*)d_in[4];
    const float* coef4 = (const float*)d_in[5];
    const float* box5  = (const float*)d_in[6];
    const float* cls5  = (const float*)d_in[7];
    const float* coef5 = (const float*)d_in[8];
    const float* proto = (const float*)d_in[9];
    const float* gtb   = (const float*)d_in[10];
    const float* gtm   = (const float*)d_in[11];
    float* out = (float*)d_out;

    float* ws = (float*)d_ws;
    size_t off = 0;
    float* boxes  = ws + off; off += (size_t)BATCH * NANCH * 4;        // 268800
    float* scores = ws + off; off += (size_t)BATCH * NANCH;            // 67200
    float* metric = ws + off; off += (size_t)BATCH * MAXGT * NANCH;    // 537600
    float* dm     = ws + off; off += (size_t)BATCH * MAXGT * HWPROTO;  // 1638400
    float* accum  = ws + off; off += BATCH * 8;
    int*   selIdx = (int*)(ws + off); off += BATCH * KSEL;
    int*   selGi  = (int*)(ws + off); off += BATCH * KSEL;
    float* selVf  = ws + off; off += BATCH * KSEL;
    _Float16* Asw = (_Float16*)(ws + off); off += (BATCH * MT * 32 * 16) / 2;
    _Float16* Bsw = (_Float16*)(ws + off); off += ((size_t)BATCH * NT * 32 * 16) / 2;  // 13.1 MB

    int totA = BATCH * NANCH;
    stage_decode<<<(totA + 255) / 256, 256, 0, stream>>>(box3, box4, box5, cls3, cls4, cls5,
                                                         boxes, scores);
    stage_assign<<<BATCH, 256, 0, stream>>>(boxes, scores, gtb, metric,
                                            selIdx, selGi, selVf, accum);
    stage_posloss<<<(BATCH * KSEL + 63) / 64, 64, 0, stream>>>(boxes, gtb, box3, box4, box5,
                                                               cls3, cls4, cls5,
                                                               selIdx, selGi, selVf, accum);
    int totD = BATCH * MAXGT * HWPROTO;
    stage_downsample<<<(totD + 255) / 256, 256, 0, stream>>>(gtm, dm);
    stage_swizzleA<<<(BATCH * MT * 32 + 63) / 64, 64, 0, stream>>>(coef3, coef4, coef5,
                                                                   selIdx, Asw);
    int totB = BATCH * NT * 32;
    stage_swizzleB<<<(totB + 255) / 256, 256, 0, stream>>>(proto, Bsw);
    int waves = BATCH * MT * (NT / TPW);    // 16000 waves, 8 per block
    stage_maskloss<<<waves / 8, 256, 0, stream>>>(Bsw, Asw, dm, selGi, selVf, accum);
    stage_final<<<1, 32, 0, stream>>>(accum, out);
}
